// CausalCoreV5_23570780520999
// MI455X (gfx1250) — matrix operand order repaired and checked by reading, not compiled, not run on hardware
//
#include <hip/hip_runtime.h>
#include <hip/hip_bf16.h>

// ---------------------------------------------------------------------------
// CausalCoreV5 on MI455X (gfx1250)
//   traj[t+1] = tanh( rot(wt) applied to {C0,S0} @ [xr,xi] )
//   C0 = Aamp*cos(raw_phase), S0 = Aamp*sin(raw_phase)   (precomputed once)
//   Weights stored as bf16 hi/lo split in WMMA *B*-fragment order (lane=(K>>3&1)*16+N,
//   elem=K) -> 128 MB/step L2-resident stream. x is carried in the *A*
//   operand (rows 0..3 = xr_hi, xi_hi, xr_lo, xi_lo), pre-swizzled to memory
//   by the combine kernel, so every WMMA input comes from a load and the
//   WMMA->VALU WAR hazard (v_nop clusters) disappears. 2 WMMAs per k-tile.
// ---------------------------------------------------------------------------

typedef __attribute__((ext_vector_type(16))) __bf16 v16bf;
typedef __attribute__((ext_vector_type(8)))  float  v8f;

#define NN        4096
#define NSTEPS    32
#define NTM       256                 // 16-row tiles per matrix
#define NTK       128                 // 32-K tiles
#define KSPLIT    8                   // K parallelism (no atomics -> deterministic)
#define TK_PER_CHUNK (NTK / KSPLIT)   // 16 k-tiles per chunk
#define FRAG_ELEMS 512                // bf16 elements per 32x16 B-fragment (1 KB)
#define XSWZ_TILE  144                // bf16 elems per x k-tile: 9 slots * 16 (288 B)

// ws layout
#define W_BYTES   ((size_t)4 * NTM * NTK * FRAG_ELEMS * sizeof(__bf16))  // 128 MiB
#define X_OFF     W_BYTES
#define X_BYTES   ((size_t)NTK * XSWZ_TILE * sizeof(__bf16))             // 36 KiB
#define U_OFF     (X_OFF + X_BYTES)

__device__ __forceinline__ size_t frag_base(int m, int tm, int tk) {
  // m: 0=C0hi 1=C0lo 2=S0hi 3=S0lo
  return (size_t)((m * NTM + tm) * NTK + tk) * FRAG_ELEMS;
}

// ---- one-time: build rotated weights directly in WMMA B-fragment layout ----
// B is KxN (32x16): lane = (K>>3&1)*16 + N, element e = K in table order; N = row within the 16-row tile.
__global__ __launch_bounds__(256) void precompute_kernel(
    const float* __restrict__ raw_S, const float* __restrict__ raw_phase,
    const float* __restrict__ raw_r, const float* __restrict__ A_mask,
    const float* __restrict__ G_gate, __bf16* __restrict__ W) {
  int idx = blockIdx.x * 256 + threadIdx.x;     // 0 .. 16M-1
  int i = idx >> 12;                            // row
  int j = idx & (NN - 1);                       // col (K dimension)
  float sig = 1.0f / (1.0f + expf(-raw_r[idx]));
  float amp = A_mask[idx] * G_gate[idx] * tanhf(raw_S[idx]) * sig;
  float ph  = raw_phase[idx];
  float c0  = amp * cosf(ph);
  float s0  = amp * sinf(ph);
  __bf16 chi = (__bf16)c0;  __bf16 clo = (__bf16)(c0 - (float)chi);
  __bf16 shi = (__bf16)s0;  __bf16 slo = (__bf16)(s0 - (float)shi);
  int tm = i >> 4, tk = j >> 5;
  size_t off = (size_t)(((j >> 3) & 1) * 16 + (i & 15)) * 16 + (size_t)((j & 7) | ((j & 16) >> 1));  // lane=(K>>3&1)*16+N, elem=K table order
  W[frag_base(0, tm, tk) + off] = chi;
  W[frag_base(1, tm, tk) + off] = clo;
  W[frag_base(2, tm, tk) + off] = shi;
  W[frag_base(3, tm, tk) + off] = slo;
}

// x swizzle: A-fragment layout, rows M=0..3 = {xr_hi, xi_hi, xr_lo, xi_lo}.
// 16-bit A 16x32 (ISA 7.12.2): lane l<16 -> M=l holds K {0..7,16..23};
// lane l+16 -> M=l holds K {8..15,24..31}. Data lanes {0..3,16..19} map to
// slots 0..7; slot 8 is a shared all-zero slot for the remaining 24 lanes.
__device__ __forceinline__ void xswz_write(__bf16* __restrict__ xswz, int i,
                                           float xr, float xi) {
  __bf16 rh = (__bf16)xr, ih = (__bf16)xi;
  __bf16 rl = (__bf16)(xr - (float)rh);
  __bf16 il = (__bf16)(xi - (float)ih);
  int tile = i >> 5, kk = i & 31;
  int hsel = (kk & 8) ? 4 : 0;                  // K-high half -> slots 4..7
  int pos  = (kk & 7) | ((kk & 16) >> 1);       // K slot within lane: 0..15
  __bf16* base = xswz + (size_t)tile * XSWZ_TILE + pos;
  base[(0 + hsel) * 16] = rh;
  base[(1 + hsel) * 16] = ih;
  base[(2 + hsel) * 16] = rl;
  base[(3 + hsel) * 16] = il;
}

// ---- one-time: traj[0] = x, swizzle x, zero the shared zero-slots ----------
__global__ __launch_bounds__(256) void init_kernel(
    const float* __restrict__ x, float* __restrict__ out,
    __bf16* __restrict__ xswz) {
  int i = blockIdx.x * 256 + threadIdx.x;
  if (i < NTK * 16) {                           // zero slot 8 of each k-tile
    xswz[(size_t)(i >> 4) * XSWZ_TILE + 8 * 16 + (i & 15)] = (__bf16)0.0f;
  }
  float xr = x[2 * i], xi = x[2 * i + 1];
  out[2 * i] = xr;  out[2 * i + 1] = xi;
  xswz_write(xswz, i, xr, xi);
}

#define WMMA_BF16(acc, a, b) \
  acc = __builtin_amdgcn_wmma_f32_16x16x32_bf16(false, a, false, b, (short)0, acc, false, false)

// ---- per step: U = [C0;S0] @ [xr,xi] via WMMA (hi/lo split, K-split) --------
__global__ __launch_bounds__(32) void matvec_kernel(
    const __bf16* __restrict__ W, const __bf16* __restrict__ xswz,
    float* __restrict__ Upart) {
  int b     = blockIdx.x;
  int g     = b & 1;                    // 0 = C0, 1 = S0
  int tm    = (b >> 1) & (NTM - 1);     // row tile
  int chunk = b >> 9;                   // K chunk
  int lane  = threadIdx.x;

  v8f acc = {};                         // rows 0..3: xr_hi, xi_hi, xr_lo, xi_lo
  const __bf16* Bhi = W + frag_base(2 * g,     tm, chunk * TK_PER_CHUNK) + lane * 16;
  const __bf16* Blo = W + frag_base(2 * g + 1, tm, chunk * TK_PER_CHUNK) + lane * 16;
  // lanes {0..3} -> slots 0..3, {16..19} -> slots 4..7, all others -> zero slot 8
  int sel = (lane < 4) ? lane : ((lane >= 16 && lane < 20) ? (lane - 12) : 8);
  const __bf16* Xp = xswz + (size_t)(chunk * TK_PER_CHUNK) * XSWZ_TILE + sel * 16;

#pragma unroll 4
  for (int it = 0; it < TK_PER_CHUNK; ++it) {
    v16bf a   = *(const v16bf*)(Xp  + (size_t)it * XSWZ_TILE);   // 32 B/lane
    v16bf bhi = *(const v16bf*)(Bhi + (size_t)it * FRAG_ELEMS);  // coalesced 1 KB
    v16bf blo = *(const v16bf*)(Blo + (size_t)it * FRAG_ELEMS);
    // acc += (x_hi + x_lo) rows  x  (W_hi + W_lo)  -> full-precision product
    WMMA_BF16(acc, a, bhi);
    WMMA_BF16(acc, a, blo);
  }

  // D layout: VGPR r, lanes 0-15 -> (M=r, N=lane). Rows: 0=xr_hi 1=xi_hi
  // 2=xr_lo 3=xi_lo; output column N = row index within the tm tile.
  if (lane < 16) {
    int row = tm * 16 + lane;
    float ur = acc[0] + acc[2];
    float ui = acc[1] + acc[3];
    Upart[(size_t)((chunk * 2 + g) * 2 + 0) * NN + row] = ur;  // vs xr
    Upart[(size_t)((chunk * 2 + g) * 2 + 1) * NN + row] = ui;  // vs xi
  }
}

// ---- per step: reduce K chunks, apply scalar rotation + tanh, emit next x ---
__global__ __launch_bounds__(256) void combine_kernel(
    const float* __restrict__ Upart, const float* __restrict__ omega_p, int t,
    float* __restrict__ out, __bf16* __restrict__ xswz) {
  int i = blockIdx.x * 256 + threadIdx.x;
  float w  = omega_p[0] * (float)t;
  float cw = cosf(w), sw = sinf(w);
  float u1 = 0.f, u2 = 0.f, u3 = 0.f, u4 = 0.f;
#pragma unroll
  for (int ch = 0; ch < KSPLIT; ++ch) {
    const float* base = Upart + (size_t)ch * 4 * NN;
    u1 += base[0 * NN + i];   // C0 @ xr
    u3 += base[1 * NN + i];   // C0 @ xi
    u2 += base[2 * NN + i];   // S0 @ xr
    u4 += base[3 * NN + i];   // S0 @ xi
  }
  // Aamp*c = cw*C0 - sw*S0 ; Aamp*s = sw*C0 + cw*S0
  float out_r = cw * u1 - sw * u2 - sw * u3 - cw * u4;
  float out_i = sw * u1 + cw * u2 + cw * u3 - sw * u4;
  float xr = tanhf(out_r), xi = tanhf(out_i);
  out[(size_t)(t + 1) * 2 * NN + 2 * i]     = xr;
  out[(size_t)(t + 1) * 2 * NN + 2 * i + 1] = xi;
  xswz_write(xswz, i, xr, xi);
}

extern "C" void kernel_launch(void* const* d_in, const int* in_sizes, int n_in,
                              void* d_out, int out_size, void* d_ws, size_t ws_size,
                              hipStream_t stream) {
  const float* x         = (const float*)d_in[0];
  const float* raw_S     = (const float*)d_in[1];
  const float* raw_phase = (const float*)d_in[2];
  const float* raw_r     = (const float*)d_in[3];
  const float* A_mask    = (const float*)d_in[4];
  const float* G_gate    = (const float*)d_in[5];
  const float* omega     = (const float*)d_in[6];
  float* out = (float*)d_out;

  char*   ws    = (char*)d_ws;
  __bf16* W     = (__bf16*)ws;
  __bf16* xswz  = (__bf16*)(ws + X_OFF);
  float*  Upart = (float*)(ws + U_OFF);

  // one-time weight rotation + fragment packing (16M threads)
  precompute_kernel<<<(NN * NN) / 256, 256, 0, stream>>>(raw_S, raw_phase, raw_r,
                                                         A_mask, G_gate, W);
  init_kernel<<<NN / 256, 256, 0, stream>>>(x, out, xswz);

  for (int t = 0; t < NSTEPS; ++t) {
    matvec_kernel<<<2 * NTM * KSPLIT, 32, 0, stream>>>(W, xswz, Upart);
    combine_kernel<<<NN / 256, 256, 0, stream>>>(Upart, omega, t, out, xswz);
  }
}
